// TemporalMoELayer_4234837754558
// MI455X (gfx1250) — compile-verified, matrix operand order checked by reading, not run on hardware
//
#include <hip/hip_runtime.h>
#include <hip/hip_bf16.h>
#include <stdint.h>

// ---------------------------------------------------------------------------
// Problem constants (reference): H=1024, I=2048, E=8. T = in_sizes[0]/H.
// ---------------------------------------------------------------------------
#define DIM_H 1024
#define DIM_I 2048
#define NUM_E 8

typedef __attribute__((ext_vector_type(16))) __bf16 v16bf;
typedef __attribute__((ext_vector_type(8)))  __bf16 v8bf;
typedef __attribute__((ext_vector_type(8)))  float  v8f;
typedef __attribute__((ext_vector_type(4)))  int    v4i;

// Async global->LDS path (gfx1250): guarded so absence degrades to sync copies.
#if defined(__HIP_DEVICE_COMPILE__) && defined(__has_builtin)
#  if __has_builtin(__builtin_amdgcn_global_load_async_to_lds_b128) && \
      __has_builtin(__builtin_amdgcn_s_wait_asynccnt)
#    define HAVE_ASYNC_LDS 1
#  endif
#endif
#ifndef HAVE_ASYNC_LDS
#  define HAVE_ASYNC_LDS 0
#endif

__device__ __forceinline__ void async_copy16(__bf16* lds_dst, const __bf16* gsrc) {
#if HAVE_ASYNC_LDS
    // signature (from clang diagnostic): (AS1 v4i*, AS3 v4i*, imm offset, imm cpol)
    __builtin_amdgcn_global_load_async_to_lds_b128(
        (__attribute__((address_space(1))) v4i*)gsrc,
        (__attribute__((address_space(3))) v4i*)lds_dst, 0, 0);
#else
    *(v8bf*)lds_dst = *(const v8bf*)gsrc;   // sync fallback (still via LDS)
#endif
}

template <int N>
__device__ __forceinline__ void wait_async() {
#if HAVE_ASYNC_LDS
    __builtin_amdgcn_s_wait_asynccnt(N);
#endif
}

// fp32 -> bf16 round-to-nearest-even (conversion kernels and SwiGLU epilogue)
__device__ __forceinline__ __bf16 f32_to_bf16(float f) {
    union { float f; uint32_t u; } v; v.f = f;
    uint32_t r = v.u + 0x7FFFu + ((v.u >> 16) & 1u);
    uint16_t h = (uint16_t)(r >> 16);
    __bf16 b;
    __builtin_memcpy(&b, &h, 2);
    return b;
}

#define WMMA_BF16(A, B, C) \
    __builtin_amdgcn_wmma_f32_16x16x32_bf16(false, (A), false, (B), (short)0, (C), false, false)

// ---------------------------------------------------------------------------
// Fragment loaders (ISA 7.12.2, wave32)
// A 16x32 (MxK) bf16 row-major: lane<16: M=lane, K={0..7,16..23};
//                               lane>=16: M=lane-16, K={8..15,24..31}
// ---------------------------------------------------------------------------
__device__ __forceinline__ v16bf load_a_rm(const __bf16* __restrict__ A, int lda,
                                           int m0, int k0, int lane) {
    const int row = lane & 15;
    const int g   = (lane >> 4) * 8;
    const __bf16* p = A + (size_t)(m0 + row) * lda + k0 + g;
    v8bf lo = *(const v8bf*)p;          // K = g .. g+7
    v8bf hi = *(const v8bf*)(p + 16);   // K = g+16 .. g+23
    return __builtin_shufflevector(lo, hi, 0,1,2,3,4,5,6,7,8,9,10,11,12,13,14,15);
}

// Same A layout, but sourced from an LDS slab laid out [row 0..63][k 0..31].
__device__ __forceinline__ v16bf lds_load_a(const __bf16* base, int m_l, int lane) {
    const int row = m_l + (lane & 15);
    const int g   = (lane >> 4) * 8;
    const __bf16* p = base + row * 32 + g;
    v8bf lo = *(const v8bf*)p;
    v8bf hi = *(const v8bf*)(p + 16);
    return __builtin_shufflevector(lo, hi, 0,1,2,3,4,5,6,7,8,9,10,11,12,13,14,15);
}

// B 32x16 (KxN) FRAGMENT-MAJOR bf16: tile (k_tile,n_tile) = 32 lanes x 16
// contiguous bf16 -> one contiguous 32B load per lane.
__device__ __forceinline__ v16bf load_b_frag(const __bf16* __restrict__ Wb,
                                             int Ktiles, int n_tile, int k_tile,
                                             int lane) {
    const __bf16* p = Wb + ((((size_t)n_tile * Ktiles + k_tile) * 32) + lane) * 16;
    return *(const v16bf*)p;
}

// ---------------------------------------------------------------------------
// Conversion kernels (cold path, bandwidth-trivial)
// ---------------------------------------------------------------------------
__global__ void cvt_rm_kernel(const float* __restrict__ in, __bf16* __restrict__ out,
                              size_t n) {
    size_t i = (size_t)blockIdx.x * blockDim.x + threadIdx.x;
    if (i < n) out[i] = f32_to_bf16(in[i]);
}

// W[K,N] fp32 row-major -> fragment-major bf16 (WMMA B layout)
__global__ void cvt_frag_kernel(const float* __restrict__ W, __bf16* __restrict__ Wb,
                                int K, int N) {
    size_t i = (size_t)blockIdx.x * blockDim.x + threadIdx.x;
    if (i >= (size_t)K * N) return;
    const int k = (int)(i / N), n = (int)(i % N);
    const int k_tile = k >> 5, kk = k & 31;
    const int n_tile = n >> 4, nn = n & 15;
    const int q    = kk >> 3;
    const int lane = nn + ((q & 1) << 4);
    const int j    = ((q >> 1) << 3) + (kk & 7);
    const int Ktiles = K >> 5;
    Wb[((((size_t)n_tile * Ktiles + k_tile) * 32) + lane) * 16 + j] = f32_to_bf16(W[i]);
}

// ---------------------------------------------------------------------------
// Epilogues
// ---------------------------------------------------------------------------
__device__ __forceinline__ void epi_xr(float* __restrict__ xr,
                                       const float* __restrict__ x,
                                       const float* __restrict__ bias,
                                       int m0, int n0, const v8f& c, int lane) {
    const int n  = lane & 15;
    const int mh = (lane >> 4) * 8;
#pragma unroll
    for (int r = 0; r < 8; ++r) {
        const int m = m0 + r + mh;
        const size_t idx = (size_t)m * DIM_H + (n0 + n);
        xr[idx] = c[r] + x[idx] + bias[n0 + n];
    }
}

__device__ __forceinline__ void epi_swiglu(__bf16* __restrict__ hbuf,
                                           int m0, int n0,
                                           const v8f& g, const v8f& u, int lane) {
    const int n  = lane & 15;
    const int mh = (lane >> 4) * 8;
#pragma unroll
    for (int r = 0; r < 8; ++r) {
        const int m = m0 + r + mh;
        const float gv = g[r];
        const float hv = (gv / (1.f + __expf(-gv))) * u[r];
        hbuf[(size_t)m * DIM_I + (n0 + n)] = f32_to_bf16(hv);
    }
}

__device__ __forceinline__ void epi_down(float* __restrict__ out,
                                         const float* __restrict__ combine,
                                         int expert, int accumulate,
                                         int m0, int n0, const v8f& c, int lane) {
    const int n  = lane & 15;
    const int mh = (lane >> 4) * 8;
#pragma unroll
    for (int r = 0; r < 8; ++r) {
        const int m = m0 + r + mh;
        const size_t idx = (size_t)m * DIM_H + (n0 + n);
        const float s = combine ? combine[(size_t)m * NUM_E + expert] : 1.0f;
        const float v = s * c[r];
        if (accumulate) out[idx] += v; else out[idx] = v;
    }
}

// ---------------------------------------------------------------------------
// xr_kernel: xr = x + tc@tp_w + tp_b.  Register-direct 32x32 per wave.
// ---------------------------------------------------------------------------
__global__ __launch_bounds__(128) void xr_kernel(const __bf16* __restrict__ tcb,
                                                 const __bf16* __restrict__ tpwb,
                                                 const float* __restrict__ bias,
                                                 const float* __restrict__ x,
                                                 float* __restrict__ xr) {
    const int lane = threadIdx.x & 31;
    const int wave = threadIdx.x >> 5;
    const int n0 = blockIdx.x * 64 + (wave & 1) * 32;
    const int m0 = blockIdx.y * 64 + (wave >> 1) * 32;
    const int Ktiles = DIM_H / 32;
    const int nt = n0 >> 4;

    v8f c00 = {}, c01 = {}, c10 = {}, c11 = {};
#pragma unroll 2
    for (int kt = 0; kt < Ktiles; ++kt) {
        v16bf a0 = load_a_rm(tcb, DIM_H, m0,      kt * 32, lane);
        v16bf a1 = load_a_rm(tcb, DIM_H, m0 + 16, kt * 32, lane);
        v16bf b0 = load_b_frag(tpwb, Ktiles, nt,     kt, lane);
        v16bf b1 = load_b_frag(tpwb, Ktiles, nt + 1, kt, lane);
        c00 = WMMA_BF16(a0, b0, c00);
        c01 = WMMA_BF16(a0, b1, c01);
        c10 = WMMA_BF16(a1, b0, c10);
        c11 = WMMA_BF16(a1, b1, c11);
    }
    epi_xr(xr, x, bias, m0,      n0,      c00, lane);
    epi_xr(xr, x, bias, m0,      n0 + 16, c01, lane);
    epi_xr(xr, x, bias, m0 + 16, n0,      c10, lane);
    epi_xr(xr, x, bias, m0 + 16, n0 + 16, c11, lane);
}

// ---------------------------------------------------------------------------
// gateup_kernel: hbuf = silu(x@Wg)*(x@Wu). Async-LDS double-buffered pipeline.
// Block = 128 thr = 4 waves, covers M64 x N64. Per K-step stage into LDS:
//   A slab 64x32 (4KB) + 4 B-fragment tiles for Wg (4KB) + 4 for Wu (4KB).
// 24 async b128 wave-ops per step -> 6 per wave (uniform ASYNCcnt).
// ---------------------------------------------------------------------------
__global__ __launch_bounds__(128) void gateup_kernel(const __bf16* __restrict__ xb,
                                                     const __bf16* __restrict__ wgb,
                                                     const __bf16* __restrict__ wub,
                                                     __bf16* __restrict__ hbuf) {
    __shared__ __align__(16) __bf16 lds_a [2][64 * 32];   // [buf][row*32 + k]
    __shared__ __align__(16) __bf16 lds_bg[2][4 * 512];   // [buf][nt_i*512 + lane*16 + j]
    __shared__ __align__(16) __bf16 lds_bu[2][4 * 512];

    const int lane = threadIdx.x & 31;
    const int wave = threadIdx.x >> 5;
    const int Ktiles = DIM_H / 32;
    const int mBlk  = blockIdx.y * 64;          // global M base of block
    const int ntBlk = blockIdx.x * 4;           // global n_tile base of block
    const int m_l   = (wave >> 1) * 32;         // wave-local M offset
    const int ntl   = (wave & 1) * 2;           // wave-local n_tile offset (of 4)
    const int n0g   = blockIdx.x * 64 + (wave & 1) * 32;
    const int m0g   = mBlk + m_l;

    // stage one K-step (kt) into buffer `buf`: 6 async 16B-per-lane ops/wave
    auto issue = [&](int buf, int kt) {
        const int cbase = wave * 6;
#pragma unroll
        for (int ci = 0; ci < 6; ++ci) {
            const int c = cbase + ci;
            const __bf16* gsrc;
            __bf16* ldst;
            if (c < 8) {                         // A slab: 8 chunks of 8 rows
                const int row = c * 8 + (lane >> 2);
                const int kp  = (lane & 3) * 8;
                gsrc = xb + (size_t)(mBlk + row) * DIM_H + kt * 32 + kp;
                ldst = &lds_a[buf][row * 32 + kp];
            } else if (c < 16) {                 // Wg tiles: 8 chunks of 512B
                const int j = c - 8, nt_i = j >> 1, half = j & 1;
                const int off = half * 256 + lane * 8;
                gsrc = wgb + ((size_t)(ntBlk + nt_i) * Ktiles + kt) * 512 + off;
                ldst = &lds_bg[buf][nt_i * 512 + off];
            } else {                             // Wu tiles: 8 chunks of 512B
                const int j = c - 16, nt_i = j >> 1, half = j & 1;
                const int off = half * 256 + lane * 8;
                gsrc = wub + ((size_t)(ntBlk + nt_i) * Ktiles + kt) * 512 + off;
                ldst = &lds_bu[buf][nt_i * 512 + off];
            }
            async_copy16(ldst, gsrc);
        }
    };

    v8f g00 = {}, g01 = {}, g10 = {}, g11 = {};
    v8f u00 = {}, u01 = {}, u10 = {}, u11 = {};

    issue(0, 0);
    int buf = 0;
    for (int kt = 0; kt < Ktiles; ++kt) {
        if (kt + 1 < Ktiles) {
            issue(buf ^ 1, kt + 1);   // prefetch next step into other buffer
            wait_async<6>();          // own current-buffer ops complete
        } else {
            wait_async<0>();
        }
        __syncthreads();              // all waves' staging visible

        v16bf a0  = lds_load_a(lds_a[buf], m_l,      lane);
        v16bf a1  = lds_load_a(lds_a[buf], m_l + 16, lane);
        v16bf bg0 = *(const v16bf*)&lds_bg[buf][(ntl + 0) * 512 + lane * 16];
        v16bf bg1 = *(const v16bf*)&lds_bg[buf][(ntl + 1) * 512 + lane * 16];
        v16bf bu0 = *(const v16bf*)&lds_bu[buf][(ntl + 0) * 512 + lane * 16];
        v16bf bu1 = *(const v16bf*)&lds_bu[buf][(ntl + 1) * 512 + lane * 16];

        g00 = WMMA_BF16(a0, bg0, g00);
        g01 = WMMA_BF16(a0, bg1, g01);
        g10 = WMMA_BF16(a1, bg0, g10);
        g11 = WMMA_BF16(a1, bg1, g11);
        u00 = WMMA_BF16(a0, bu0, u00);
        u01 = WMMA_BF16(a0, bu1, u01);
        u10 = WMMA_BF16(a1, bu0, u10);
        u11 = WMMA_BF16(a1, bu1, u11);

        __syncthreads();              // done reading before buffer reuse
        buf ^= 1;
    }

    epi_swiglu(hbuf, m0g,      n0g,      g00, u00, lane);
    epi_swiglu(hbuf, m0g,      n0g + 16, g01, u01, lane);
    epi_swiglu(hbuf, m0g + 16, n0g,      g10, u10, lane);
    epi_swiglu(hbuf, m0g + 16, n0g + 16, g11, u11, lane);
}

// ---------------------------------------------------------------------------
// down_kernel: out (+)= combine[t,e] * (hbuf @ Wd). Register-direct 32x32/wave.
// ---------------------------------------------------------------------------
__global__ __launch_bounds__(128) void down_kernel(const __bf16* __restrict__ hbuf,
                                                   const __bf16* __restrict__ wdb,
                                                   const float* __restrict__ combine,
                                                   int expert,
                                                   float* __restrict__ out,
                                                   int accumulate) {
    const int lane = threadIdx.x & 31;
    const int wave = threadIdx.x >> 5;
    const int n0 = blockIdx.x * 64 + (wave & 1) * 32;
    const int m0 = blockIdx.y * 64 + (wave >> 1) * 32;
    const int Ktiles = DIM_I / 32;
    const int nt = n0 >> 4;

    v8f c00 = {}, c01 = {}, c10 = {}, c11 = {};
#pragma unroll 2
    for (int kt = 0; kt < Ktiles; ++kt) {
        v16bf a0 = load_a_rm(hbuf, DIM_I, m0,      kt * 32, lane);
        v16bf a1 = load_a_rm(hbuf, DIM_I, m0 + 16, kt * 32, lane);
        v16bf b0 = load_b_frag(wdb, Ktiles, nt,     kt, lane);
        v16bf b1 = load_b_frag(wdb, Ktiles, nt + 1, kt, lane);
        c00 = WMMA_BF16(a0, b0, c00);
        c01 = WMMA_BF16(a0, b1, c01);
        c10 = WMMA_BF16(a1, b0, c10);
        c11 = WMMA_BF16(a1, b1, c11);
    }
    epi_down(out, combine, expert, accumulate, m0,      n0,      c00, lane);
    epi_down(out, combine, expert, accumulate, m0,      n0 + 16, c01, lane);
    epi_down(out, combine, expert, accumulate, m0 + 16, n0,      c10, lane);
    epi_down(out, combine, expert, accumulate, m0 + 16, n0 + 16, c11, lane);
}

// ---------------------------------------------------------------------------
// Router: logits = xr @ gate_w, softmax, top-k, renorm -> dense combine[T,E]
// ---------------------------------------------------------------------------
__global__ void router_kernel(const float* __restrict__ xr,
                              const float* __restrict__ gate_w,
                              float* __restrict__ combine,
                              const int* __restrict__ topk_ptr, int T) {
    const int t = blockIdx.x * blockDim.x + threadIdx.x;
    if (t >= T) return;
    const float* xrow = xr + (size_t)t * DIM_H;

    float logits[NUM_E];
#pragma unroll
    for (int e = 0; e < NUM_E; ++e) logits[e] = 0.f;
    for (int h = 0; h < DIM_H; ++h) {
        const float xv = xrow[h];
#pragma unroll
        for (int e = 0; e < NUM_E; ++e)
            logits[e] = fmaf(xv, gate_w[(size_t)h * NUM_E + e], logits[e]);
    }
    float mx = logits[0];
#pragma unroll
    for (int e = 1; e < NUM_E; ++e) mx = fmaxf(mx, logits[e]);
    float probs[NUM_E]; float sum = 0.f;
#pragma unroll
    for (int e = 0; e < NUM_E; ++e) { probs[e] = __expf(logits[e] - mx); sum += probs[e]; }
    const float invs = 1.f / sum;
#pragma unroll
    for (int e = 0; e < NUM_E; ++e) probs[e] *= invs;

    int K = *topk_ptr; if (K > NUM_E) K = NUM_E; if (K < 1) K = 1;
    float w[NUM_E]; bool used[NUM_E];
#pragma unroll
    for (int e = 0; e < NUM_E; ++e) { w[e] = 0.f; used[e] = false; }
    float psum = 0.f;
    for (int k = 0; k < K; ++k) {
        int best = 0; float bv = -1.f;
#pragma unroll
        for (int e = 0; e < NUM_E; ++e)
            if (!used[e] && probs[e] > bv) { bv = probs[e]; best = e; }
        used[best] = true; w[best] = bv; psum += bv;
    }
    const float inv = 1.f / (psum + 1e-5f);
#pragma unroll
    for (int e = 0; e < NUM_E; ++e) combine[(size_t)t * NUM_E + e] = w[e] * inv;
}

// ---------------------------------------------------------------------------
// kernel_launch — inputs (setup_inputs order):
//  0:x 1:temporal_context 2:tp_w[H,H] 3:tp_b[H] 4:gate_w[H,E]
//  5:eg[E,H,I] 6:eu[E,H,I] 7:ed[E,I,H] 8:sg 9:su 10:sd 11:top_k
// ---------------------------------------------------------------------------
extern "C" void kernel_launch(void* const* d_in, const int* in_sizes, int n_in,
                              void* d_out, int out_size, void* d_ws, size_t ws_size,
                              hipStream_t stream) {
    const float* x      = (const float*)d_in[0];
    const float* tc     = (const float*)d_in[1];
    const float* tp_w   = (const float*)d_in[2];
    const float* tp_b   = (const float*)d_in[3];
    const float* gate_w = (const float*)d_in[4];
    const float* eg     = (const float*)d_in[5];
    const float* eu     = (const float*)d_in[6];
    const float* ed     = (const float*)d_in[7];
    const float* sg     = (const float*)d_in[8];
    const float* su     = (const float*)d_in[9];
    const float* sd     = (const float*)d_in[10];
    const int*   topk   = (const int*)d_in[11];
    float*       out    = (float*)d_out;

    const int T = in_sizes[0] / DIM_H;   // 4096

    // ---- workspace carve-out (256B aligned) ----
    char* ws = (char*)d_ws;
    size_t off = 0;
    auto carve = [&](size_t bytes) {
        char* p = ws + off;
        off += (bytes + 255) & ~(size_t)255;
        return p;
    };
    float*  xr      = (float*)carve((size_t)T * DIM_H * 4);
    float*  combine = (float*)carve((size_t)T * NUM_E * 4);
    __bf16* xb      = (__bf16*)carve((size_t)T * DIM_H * 2);
    __bf16* tcb     = (__bf16*)carve((size_t)T * DIM_H * 2);
    __bf16* hbuf    = (__bf16*)carve((size_t)T * DIM_I * 2);
    __bf16* tpwb    = (__bf16*)carve((size_t)DIM_H * DIM_H * 2);
    __bf16* wgb     = (__bf16*)carve((size_t)DIM_H * DIM_I * 2);
    __bf16* wub     = (__bf16*)carve((size_t)DIM_H * DIM_I * 2);
    __bf16* wdb     = (__bf16*)carve((size_t)DIM_I * DIM_H * 2);

    const dim3 blk(128);
    const dim3 gH(DIM_H / 64, (unsigned)(T / 64));
    const dim3 gI(DIM_I / 64, (unsigned)(T / 64));
    const int CVT = 256;
    const size_t nTH = (size_t)T * DIM_H;
    const size_t nHH = (size_t)DIM_H * DIM_H;
    const size_t nHI = (size_t)DIM_H * DIM_I;
    const size_t nIH = (size_t)DIM_I * DIM_H;

    // 0) one-time operand conversions
    cvt_rm_kernel<<<dim3((nTH + CVT - 1) / CVT), CVT, 0, stream>>>(x,  xb,  nTH);
    cvt_rm_kernel<<<dim3((nTH + CVT - 1) / CVT), CVT, 0, stream>>>(tc, tcb, nTH);
    cvt_frag_kernel<<<dim3((nHH + CVT - 1) / CVT), CVT, 0, stream>>>(tp_w, tpwb, DIM_H, DIM_H);

    // 1) temporal projection + residual + bias
    xr_kernel<<<gH, blk, 0, stream>>>(tcb, tpwb, tp_b, x, xr);

    // 2) router -> dense combine weights
    router_kernel<<<dim3((T + 127) / 128), 128, 0, stream>>>(xr, gate_w, combine, topk, T);

    // 3) shared expert (write mode initializes d_out)
    cvt_frag_kernel<<<dim3((nHI + CVT - 1) / CVT), CVT, 0, stream>>>(sg, wgb, DIM_H, DIM_I);
    cvt_frag_kernel<<<dim3((nHI + CVT - 1) / CVT), CVT, 0, stream>>>(su, wub, DIM_H, DIM_I);
    gateup_kernel<<<gI, blk, 0, stream>>>(xb, wgb, wub, hbuf);
    cvt_frag_kernel<<<dim3((nIH + CVT - 1) / CVT), CVT, 0, stream>>>(sd, wdb, DIM_I, DIM_H);
    down_kernel<<<gH, blk, 0, stream>>>(hbuf, wdb, nullptr, 0, out, /*accumulate=*/0);

    // 4) routed experts, accumulated with combine weights
    for (int e = 0; e < NUM_E; ++e) {
        cvt_frag_kernel<<<dim3((nHI + CVT - 1) / CVT), CVT, 0, stream>>>(
            eg + (size_t)e * nHI, wgb, DIM_H, DIM_I);
        cvt_frag_kernel<<<dim3((nHI + CVT - 1) / CVT), CVT, 0, stream>>>(
            eu + (size_t)e * nHI, wub, DIM_H, DIM_I);
        gateup_kernel<<<gI, blk, 0, stream>>>(xb, wgb, wub, hbuf);
        cvt_frag_kernel<<<dim3((nIH + CVT - 1) / CVT), CVT, 0, stream>>>(
            ed + (size_t)e * nIH, wdb, DIM_I, DIM_H);
        down_kernel<<<gH, blk, 0, stream>>>(hbuf, wdb, combine, e, out, /*accumulate=*/1);
    }
    (void)n_in; (void)out_size; (void)ws_size;
}
